// WindowAttention_69037304316425
// MI455X (gfx1250) — matrix operand order repaired and checked by reading, not compiled, not run on hardware
//
#include <hip/hip_runtime.h>

typedef __attribute__((ext_vector_type(16))) _Float16 v16h;
typedef __attribute__((ext_vector_type(8)))  _Float16 v8h;
typedef __attribute__((ext_vector_type(8)))  float    v8f;

#define WN      98      // tokens per window
#define NP      112     // padded tokens (7 * 16)
#define EMBED   192
#define C3      576
#define HEADS   6
#define HD      32
#define NW      64
#define XS      200     // half-stride Xh/Qh/Kh/Oh: 400B rows, banks r*36%64 distinct
#define VS      136     // half-stride Vt/Sb/Pb: 272B rows, banks r*4%64 distinct
#define SCALE   0.17677669529663687f   // 32^-0.5

// ---- WMMA fragment loaders (layouts per CDNA5 ISA 7.12.2, wave32) ----
__device__ __forceinline__ v16h frag_a(const _Float16* p, int stride, int row0,
                                       int k0, int lane) {
  int r = lane & 15, hi = lane >> 4;
  int off = (row0 + r) * stride + k0 + hi * 8;
  v8h lo = *(const v8h*)(p + off);
  v8h hh = *(const v8h*)(p + off + 16);
  return __builtin_shufflevector(lo, hh, 0,1,2,3,4,5,6,7,8,9,10,11,12,13,14,15);
}
__device__ __forceinline__ v16h frag_b(const _Float16* p, int stride, int n0,
                                       int k0, int lane) {
  int n = lane & 15, hi = lane >> 4;
  int off = (n0 + n) * stride + k0 + hi * 8;
  v8h lo = *(const v8h*)(p + off);
  v8h hh = *(const v8h*)(p + off + 16);
  return __builtin_shufflevector(lo, hh, 0,1,2,3,4,5,6,7,8,9,10,11,12,13,14,15);
}
__device__ __forceinline__ v8f wmma16(v16h a, v16h b, v8f c) {
  return __builtin_amdgcn_wmma_f32_16x16x32_f16(false, a, false, b, (short)0, c,
                                                false, false);
}

// ---- prep: fp32->f16 weights, gather relative-position bias to [H][98][98] ----
__global__ void wattn_prep(const float* __restrict__ qkv_w,
                           const float* __restrict__ o_w,
                           const float* __restrict__ bias_table,
                           const int*   __restrict__ rel_index,
                           _Float16* __restrict__ qkvWh,
                           _Float16* __restrict__ oWh,
                           float* __restrict__ biasF) {
  int idx = blockIdx.x * blockDim.x + threadIdx.x;
  const int n_qkv = C3 * EMBED;          // 110592
  const int n_ow  = EMBED * EMBED;       // 36864
  const int n_b   = HEADS * WN * WN;     // 57624
  if (idx < n_qkv) {
    qkvWh[idx] = (_Float16)qkv_w[idx];
  } else if (idx < n_qkv + n_ow) {
    int j = idx - n_qkv;
    oWh[j] = (_Float16)o_w[j];
  } else if (idx < n_qkv + n_ow + n_b) {
    int j = idx - n_qkv - n_ow;
    int h = j / (WN * WN);
    int rem = j % (WN * WN);
    biasF[j] = bias_table[rel_index[rem] * HEADS + h];
  }
}

// ---- fused window attention: one block per window ----
__launch_bounds__(256)
__global__ void wattn_main(const float* __restrict__ x,
                           const float* __restrict__ mask,
                           const float* __restrict__ qkv_b,
                           const float* __restrict__ o_b,
                           const _Float16* __restrict__ qkvWh,
                           const _Float16* __restrict__ oWh,
                           const float* __restrict__ biasF,
                           float* __restrict__ out) {
  // Xh (phase 1) overlaid with Sb+Pb (phases 2-3): disjoint lifetimes.
  __shared__ __align__(16) char uni[NP * VS * 4 + NP * VS * 2];  // 91392 B
  __shared__ __align__(16) _Float16 Qh[NP * XS];    // scaled Q  [tok][c]
  __shared__ __align__(16) _Float16 Kh[NP * XS];    // K         [tok][c]
  __shared__ __align__(16) _Float16 Vt[EMBED * VS]; // V^T       [c][tok]
  __shared__ __align__(16) _Float16 Oh[NP * XS];    // attn out  [tok][c]
  _Float16* Xh = (_Float16*)uni;
  float*    Sb = (float*)uni;                       // [NP][VS] fp32 scores
  _Float16* Pb = (_Float16*)(uni + NP * VS * 4);    // [NP][VS] f16 probs

  const int tid  = threadIdx.x;
  const int lane = tid & 31;
  const int wv   = tid >> 5;
  const int nl   = lane & 15;
  const int hi   = lane >> 4;
  const size_t xbase = (size_t)blockIdx.x * WN * EMBED;

  // ---- phase 0: stage x as f16, zero pads that feed WMMA ----
  for (int i = tid; i < NP * EMBED; i += 256) {
    int r = i / EMBED, c = i % EMBED;
    float v = (r < WN) ? x[xbase + (size_t)r * EMBED + c] : 0.f;
    Xh[r * XS + c] = (_Float16)v;
  }
  for (int i = tid; i < EMBED * (VS - NP); i += 256) {  // Vt pad key cols
    int r = i / (VS - NP), c = NP + i % (VS - NP);
    Vt[r * VS + c] = (_Float16)0.f;
  }
  __syncthreads();

  // ---- phase 1: QKV = Xh(112x192) @ qkv_w^T(192x576) + b ----
  // A-fragment reused across 3 N-tiles; group of 48 cols never straddles the
  // Q/K/V boundaries (192, 384 are multiples of 48) -> uniform routing.
  for (int g = wv; g < 7 * 12; g += 8) {
    int mt = g % 7, ng = g / 7;
    int nb = ng * 48;
    v8f a0 = {}, a1 = {}, a2 = {};
    for (int kt = 0; kt < 6; ++kt) {
      v16h a = frag_a(Xh, XS, mt * 16, kt * 32, lane);
      a0 = wmma16(a, frag_b(qkvWh, EMBED, nb,      kt * 32, lane), a0);
      a1 = wmma16(a, frag_b(qkvWh, EMBED, nb + 16, kt * 32, lane), a1);
      a2 = wmma16(a, frag_b(qkvWh, EMBED, nb + 32, kt * 32, lane), a2);
    }
    #pragma unroll
    for (int j = 0; j < 3; ++j) {
      v8f acc = (j == 0) ? a0 : (j == 1) ? a1 : a2;
      int n = nb + j * 16 + nl;
      float bv = qkv_b[n];
      if (ng < 4) {              // Q (fold in 1/sqrt(hd))
        #pragma unroll
        for (int r = 0; r < 8; ++r) {
          int m = mt * 16 + r + 8 * hi;
          Qh[m * XS + n] = (_Float16)((acc[r] + bv) * SCALE);
        }
      } else if (ng < 8) {       // K
        #pragma unroll
        for (int r = 0; r < 8; ++r) {
          int m = mt * 16 + r + 8 * hi;
          Kh[m * XS + (n - EMBED)] = (_Float16)(acc[r] + bv);
        }
      } else {                   // V, transposed to [c][tok]
        #pragma unroll
        for (int r = 0; r < 8; ++r) {
          int m = mt * 16 + r + 8 * hi;
          Vt[(n - 2 * EMBED) * VS + m] = (_Float16)(acc[r] + bv);
        }
      }
    }
  }
  __syncthreads();

  const float* mrow = mask + (size_t)(blockIdx.x % NW) * WN * WN;

  // ---- phases 2-3 per head: S = QK^T, softmax(+bias+mask), O = P V ----
  for (int h = 0; h < HEADS; ++h) {
    // S: one M-row of tiles per wave, A fragment reused across all 7 N-tiles
    if (wv < 7) {
      int mt = wv;
      v16h a = frag_a(Qh + h * HD, XS, mt * 16, 0, lane);
      for (int nt = 0; nt < 7; ++nt) {
        v16h bb = frag_b(Kh + h * HD, XS, nt * 16, 0, lane);
        v8f acc = {};
        acc = wmma16(a, bb, acc);
        #pragma unroll
        for (int r = 0; r < 8; ++r)
          Sb[(mt * 16 + r + 8 * hi) * VS + nt * 16 + nl] = acc[r];
      }
    }
    __syncthreads();

    // row softmax with rel-pos bias + window mask; write P f16 (pads zeroed)
    const float* brow = biasF + h * WN * WN;
    for (int row = tid; row < NP; row += 256) {
      float*    srow = Sb + row * VS;
      _Float16* prow = Pb + row * VS;
      float mx = -3.0e38f;
      if (row < WN) {
        const float* mp = mrow + row * WN;
        const float* bp = brow + row * WN;
        for (int j = 0; j < WN; ++j) {
          float v = srow[j] + bp[j] + mp[j];
          srow[j] = v;
          mx = fmaxf(mx, v);
        }
      } else {
        for (int j = 0; j < WN; ++j) mx = fmaxf(mx, srow[j]);
      }
      float s = 0.f;
      for (int j = 0; j < WN; ++j) {
        float e = __expf(srow[j] - mx);
        srow[j] = e;
        s += e;
      }
      float rinv = 1.f / s;
      for (int j = 0; j < WN; ++j) prow[j] = (_Float16)(srow[j] * rinv);
      for (int j = WN; j < VS; ++j) prow[j] = (_Float16)0.f;
    }
    __syncthreads();

    // O_h = P(112x112, K padded to 128) @ V_h(112x32): A reused across 2 N-tiles
    if (wv < 7) {
      int mt = wv;
      v8f a0 = {}, a1 = {};
      for (int kt = 0; kt < 4; ++kt) {
        v16h a = frag_a(Pb, VS, mt * 16, kt * 32, lane);
        a0 = wmma16(a, frag_b(Vt + h * HD * VS, VS, 0,  kt * 32, lane), a0);
        a1 = wmma16(a, frag_b(Vt + h * HD * VS, VS, 16, kt * 32, lane), a1);
      }
      #pragma unroll
      for (int r = 0; r < 8; ++r) {
        int m = mt * 16 + r + 8 * hi;
        Oh[m * XS + h * HD + nl]      = (_Float16)a0[r];
        Oh[m * XS + h * HD + 16 + nl] = (_Float16)a1[r];
      }
    }
    __syncthreads();
  }

  // ---- phase 4: out = Oh(112x192) @ o_w^T(192x192) + o_b ----
  for (int g = wv; g < 7 * 6; g += 8) {
    int mt = g % 7, np2 = g / 7;
    int n0 = np2 * 32;
    v8f a0 = {}, a1 = {};
    for (int kt = 0; kt < 6; ++kt) {
      v16h a = frag_a(Oh, XS, mt * 16, kt * 32, lane);
      a0 = wmma16(a, frag_b(oWh, EMBED, n0,      kt * 32, lane), a0);
      a1 = wmma16(a, frag_b(oWh, EMBED, n0 + 16, kt * 32, lane), a1);
    }
    float ob0 = o_b[n0 + nl];
    float ob1 = o_b[n0 + 16 + nl];
    #pragma unroll
    for (int r = 0; r < 8; ++r) {
      int m = mt * 16 + r + 8 * hi;
      if (m < WN) {
        out[xbase + (size_t)m * EMBED + n0 + nl]      = a0[r] + ob0;
        out[xbase + (size_t)m * EMBED + n0 + 16 + nl] = a1[r] + ob1;
      }
    }
  }
}

extern "C" void kernel_launch(void* const* d_in, const int* in_sizes, int n_in,
                              void* d_out, int out_size, void* d_ws,
                              size_t ws_size, hipStream_t stream) {
  const float* x          = (const float*)d_in[0];
  const float* mask       = (const float*)d_in[1];
  const float* qkv_w      = (const float*)d_in[2];
  const float* qkv_b      = (const float*)d_in[3];
  const float* o_w        = (const float*)d_in[4];
  const float* o_b        = (const float*)d_in[5];
  const float* bias_table = (const float*)d_in[6];
  const int*   rel_index  = (const int*)d_in[7];
  float* out = (float*)d_out;

  char* ws = (char*)d_ws;
  _Float16* qkvWh = (_Float16*)ws;                        // 576*192 f16
  _Float16* oWh   = qkvWh + C3 * EMBED;                   // 192*192 f16
  float*    biasF = (float*)(ws + (C3 * EMBED + EMBED * EMBED) * 2);  // 6*98*98 f32

  int B = in_sizes[0] / (WN * EMBED);                     // 2048
  int prep_total = C3 * EMBED + EMBED * EMBED + HEADS * WN * WN;

  wattn_prep<<<(prep_total + 255) / 256, 256, 0, stream>>>(
      qkv_w, o_w, bias_table, rel_index, qkvWh, oWh, biasF);
  wattn_main<<<B, 256, 0, stream>>>(x, mask, qkv_b, o_b, qkvWh, oWh, biasF, out);
}
